// VQ_59803124629529
// MI455X (gfx1250) — matrix operand order repaired and checked by reading, not compiled, not run on hardware
//
#include <hip/hip_runtime.h>
#include <hip/hip_bf16.h>

// ---------------------------------------------------------------------------
// VQ-VAE vector quantization for MI455X (gfx1250, wave32, WMMA).
//   x: [131072, 64] fp32, emb: [512, 64] fp32
//   S = X @ E^T via v_wmma_f32_16x16x32_bf16 (Kred=64 -> 2 WMMA per tile),
//   B fragments double-buffered from LDS to keep the matrix pipe unstalled.
//   argmin_k (||e_k||^2 - 2*S[n,k]);  out = emb[idx] (exact fp32)
//   loss = 3.0 * mean((q - x)^2)  -> d_out[N*D]
// ---------------------------------------------------------------------------

typedef __attribute__((ext_vector_type(16))) __bf16 v16bf;
typedef __attribute__((ext_vector_type(8)))  __bf16 v8bf;
typedef __attribute__((ext_vector_type(8)))  float  v8f;

#define DIM 64
#define KCODES 512
#define HALF_CODES 256
#define ROWS_PER_BLOCK 128
#define LDS_STRIDE 72   // padded bf16 row stride (144 B): conflict-free, 16B aligned

union F16x16 { v16bf v; unsigned short u[16]; };

__device__ __forceinline__ unsigned short f2bf(float f) {
  unsigned int u = __float_as_uint(f);
  u += 0x7fffu + ((u >> 16) & 1u);   // round-to-nearest-even
  return (unsigned short)(u >> 16);
}

__device__ __forceinline__ void cv4(unsigned short* d, float4 v) {
  d[0] = f2bf(v.x); d[1] = f2bf(v.y); d[2] = f2bf(v.z); d[3] = f2bf(v.w);
}

__global__ void vq_zero(float* ws) { ws[0] = 0.0f; }

__global__ void vq_finalize(const float* __restrict__ ws, float* __restrict__ out, int nd) {
  out[nd] = 3.0f * ws[0] / (float)nd;   // codebook + BETA(=2)*commitment
}

__global__ __launch_bounds__(256)
void vq_main(const float* __restrict__ x, const float* __restrict__ emb,
             float* __restrict__ out, float* __restrict__ wsum) {
  __shared__ alignas(16) unsigned short Elds[HALF_CODES * LDS_STRIDE];
  __shared__ float e2[KCODES];
  __shared__ int   idxLds[ROWS_PER_BLOCK];
  __shared__ float wsse[8];

  const int tid  = threadIdx.x;
  const int lane = tid & 31;
  const int wave = tid >> 5;
  const int rowBase = blockIdx.x * ROWS_PER_BLOCK + wave * 16;

  // ||e_k||^2 in fp32 for all 512 codes (tiny; hits L2/WGP$)
  for (int k = tid; k < KCODES; k += 256) {
    const float* er = emb + k * DIM;
    float s = 0.f;
    #pragma unroll 8
    for (int d = 0; d < DIM; ++d) { float v = er[d]; s = fmaf(v, v, s); }
    e2[k] = s;
  }

  // -------- A fragments: this wave's 16 rows, bf16, ISA 16-bit 16x32 layout.
  // lane<16: row=lane,  K elems {0..7, 16..23}(+chunk*32)
  // lane>=16: row=lane-16, K elems {8..15, 24..31}(+chunk*32)
  const int mrow  = lane & 15;
  const int dbase = (lane < 16) ? 0 : 8;
  const float* xrow = x + (size_t)(rowBase + mrow) * DIM;
  F16x16 a0, a1;
  {
    float4 p0 = *(const float4*)(xrow + dbase);
    float4 p1 = *(const float4*)(xrow + dbase + 4);
    float4 p2 = *(const float4*)(xrow + dbase + 16);
    float4 p3 = *(const float4*)(xrow + dbase + 20);
    cv4(&a0.u[0], p0); cv4(&a0.u[4], p1); cv4(&a0.u[8], p2); cv4(&a0.u[12], p3);
    float4 q0 = *(const float4*)(xrow + dbase + 32);
    float4 q1 = *(const float4*)(xrow + dbase + 36);
    float4 q2 = *(const float4*)(xrow + dbase + 48);
    float4 q3 = *(const float4*)(xrow + dbase + 52);
    cv4(&a1.u[0], q0); cv4(&a1.u[4], q1); cv4(&a1.u[8], q2); cv4(&a1.u[12], q3);
  }

  float best[8];
  int   bidx[8];
  #pragma unroll
  for (int r = 0; r < 8; ++r) { best[r] = __builtin_inff(); bidx[r] = 0; }

  const int dB    = (lane < 16) ? 0 : 8;  // B layout: lanes 0-15 hold K {0..7,16..23}, 16-31 hold {8..15,24..31}
  const int laneC = lane & 15;            // code column within a 16-wide tile

  for (int h = 0; h < 2; ++h) {
    __syncthreads();   // protect Elds reuse across halves (also fences e2 on h==0)
    // Convert 256 codes of the codebook to bf16 in LDS (padded stride).
    for (int i = tid; i < HALF_CODES * DIM; i += 256) {
      int k = i >> 6, d = i & 63;
      Elds[k * LDS_STRIDE + d] = f2bf(emb[(h * HALF_CODES + k) * DIM + d]);
    }
    __syncthreads();

    // ---- software-pipelined tile loop: prefetch B(t+1) while computing t ----
    F16x16 b0c, b1c;
    float  e2c;
    {
      const unsigned short* ep = &Elds[laneC * LDS_STRIDE + dB];
      b0c.v = __builtin_shufflevector(*(const v8bf*)ep, *(const v8bf*)(ep + 16),
                                      0,1,2,3,4,5,6,7,8,9,10,11,12,13,14,15);
      b1c.v = __builtin_shufflevector(*(const v8bf*)(ep + 32), *(const v8bf*)(ep + 48),
                                      0,1,2,3,4,5,6,7,8,9,10,11,12,13,14,15);
      e2c   = e2[h * HALF_CODES + laneC];
    }

    for (int t = 0; t < 16; ++t) {
      // branchless prefetch of the next tile ((t+1)&15 wraps harmlessly)
      const int tn = (t + 1) & 15;
      const unsigned short* epn = &Elds[(tn * 16 + laneC) * LDS_STRIDE + dB];
      F16x16 b0n, b1n;
      b0n.v = __builtin_shufflevector(*(const v8bf*)epn, *(const v8bf*)(epn + 16),
                                      0,1,2,3,4,5,6,7,8,9,10,11,12,13,14,15);
      b1n.v = __builtin_shufflevector(*(const v8bf*)(epn + 32), *(const v8bf*)(epn + 48),
                                      0,1,2,3,4,5,6,7,8,9,10,11,12,13,14,15);
      const float e2n = e2[h * HALF_CODES + tn * 16 + laneC];

      v8f c = {0.f,0.f,0.f,0.f,0.f,0.f,0.f,0.f};
      c = __builtin_amdgcn_wmma_f32_16x16x32_bf16(false, a0.v, false, b0c.v,
                                                  (short)0, c, false, false);
      c = __builtin_amdgcn_wmma_f32_16x16x32_bf16(false, a1.v, false, b1c.v,
                                                  (short)0, c, false, false);

      const int code = h * HALF_CODES + t * 16 + laneC;
      #pragma unroll
      for (int r = 0; r < 8; ++r) {
        float s = fmaf(-2.f, c[r], e2c);     // ||x||^2 constant per row: dropped
        if (s < best[r]) { best[r] = s; bidx[r] = code; }
      }

      b0c = b0n; b1c = b1n; e2c = e2n;
    }
  }

  // Cross-lane argmin within each 16-lane half (C layout: lane = code column).
  #pragma unroll
  for (int off = 1; off <= 8; off <<= 1) {
    #pragma unroll
    for (int r = 0; r < 8; ++r) {
      float os = __shfl_xor(best[r], off, 32);
      int   oi = __shfl_xor(bidx[r], off, 32);
      if (os < best[r] || (os == best[r] && oi < bidx[r])) { best[r] = os; bidx[r] = oi; }
    }
  }

  // lane 0 holds rows 0..7 (VGPR r = row r); lane 16 holds rows 8..15
  if ((lane & 15) == 0) {
    int base = wave * 16 + ((lane >> 4) << 3);
    #pragma unroll
    for (int r = 0; r < 8; ++r) idxLds[base + r] = bidx[r];
  }
  __syncthreads();

  // Gather exact fp32 embedding rows, write q, accumulate SSE.
  float sse = 0.f;
  for (int m = 0; m < 16; ++m) {
    const int row  = rowBase + m;
    const int code = idxLds[wave * 16 + m];
    const float2 e  = ((const float2*)(emb + (size_t)code * DIM))[lane];
    const float2 xv = ((const float2*)(x   + (size_t)row  * DIM))[lane];
    ((float2*)(out + (size_t)row * DIM))[lane] = e;
    float dx = e.x - xv.x, dy = e.y - xv.y;
    sse = fmaf(dx, dx, sse);
    sse = fmaf(dy, dy, sse);
  }

  #pragma unroll
  for (int off = 1; off < 32; off <<= 1) sse += __shfl_xor(sse, off, 32);
  if (lane == 0) wsse[wave] = sse;
  __syncthreads();
  if (tid == 0) {
    float s = 0.f;
    #pragma unroll
    for (int w = 0; w < 8; ++w) s += wsse[w];
    atomicAdd(wsum, s);
  }
}

extern "C" void kernel_launch(void* const* d_in, const int* in_sizes, int n_in,
                              void* d_out, int out_size, void* d_ws, size_t ws_size,
                              hipStream_t stream) {
  (void)n_in; (void)out_size; (void)ws_size;
  const float* x   = (const float*)d_in[0];
  const float* emb = (const float*)d_in[1];
  float* out = (float*)d_out;
  float* ws  = (float*)d_ws;

  const int nd     = in_sizes[0];             // 8388608 elements of x / out tensor
  const int nrows  = nd / DIM;                // 131072
  const int blocks = nrows / ROWS_PER_BLOCK;  // 1024

  vq_zero<<<1, 1, 0, stream>>>(ws);
  vq_main<<<blocks, 256, 0, stream>>>(x, emb, out, ws);
  vq_finalize<<<1, 1, 0, stream>>>(ws, out, nd);
}
